// SurfaceVAE_FSQ_43550968382251
// MI455X (gfx1250) — matrix operand order repaired and checked by reading, not compiled, not run on hardware
//
#include <hip/hip_runtime.h>
#include <hip/hip_bf16.h>
#include <math.h>

typedef __attribute__((ext_vector_type(16))) _Float16 v16h;
typedef __attribute__((ext_vector_type(8)))  _Float16 v8h;
typedef __attribute__((ext_vector_type(8)))  float    v8f;
typedef unsigned int u32x4 __attribute__((ext_vector_type(4)));
typedef int          i32x8 __attribute__((ext_vector_type(8)));
typedef int          i32x4 __attribute__((ext_vector_type(4)));

#define WAVES_PER_BLOCK 2
#define ROWS_PER_WAVE   16
#define PANEL_HALVES    2048            // 4 KB per panel buffer
#define PANEL_BYTES     (PANEL_HALVES * 2)

// ---------------------------------------------------------------------------
// Weight packing: f32 (K x N) row-major -> f16 WMMA-B fragment order.
// For each (kt, nt) 32x16 tile: lane holds column nt*16+(lane&15), slot s holds
// k = kt*32 + 16*(s>>3) + 8*(lane>>4) + (s&7). Linear: out[((nt*Kt+kt)*32+lane)*16+slot].
// ---------------------------------------------------------------------------
__global__ void pack_weights(const float* __restrict__ W, _Float16* __restrict__ out,
                             int K, int Kp, int N) {
  int o = blockIdx.x * blockDim.x + threadIdx.x;
  int total = Kp * N;
  if (o >= total) return;
  int Kt   = Kp >> 5;
  int slot = o & 15;
  int lane = (o >> 4) & 31;
  int kt   = (o >> 9) % Kt;
  int nt   = (o >> 9) / Kt;
  int k = (kt << 5) + ((slot >> 3) << 4) + ((lane >> 4) << 3) + (slot & 7);
  int n = (nt << 4) + (lane & 15);
  float v = (k < K) ? W[(size_t)k * N + n] : 0.0f;
  out[o] = (_Float16)v;
}

// ---------------------------------------------------------------------------
// Device helpers
// ---------------------------------------------------------------------------
__device__ __forceinline__ void lds_fence() {
  asm volatile("s_wait_dscnt 0" ::: "memory");
}

// TDM: 1-D async DMA of `nbytes` (multiple of 8) from global to LDS.
// D# built per CDNA5 ISA: count=1, type=2("image"),
// data_size=3 (8B units), 1-D tile (tile_dim1..4 = 0 => unused).
// 6-arg builtin form (clang-23 / therock headers).
__device__ __forceinline__ void tdm_copy_to_lds(const void* gsrc,
                                                unsigned lds_off,
                                                unsigned nbytes) {
  unsigned long long ga = (unsigned long long)(uintptr_t)gsrc;
  unsigned nelem = nbytes >> 3;  // 8-byte units
  u32x4 g0;
  g0[0] = 1u;                                              // count=1 (user mode)
  g0[1] = lds_off;                                         // lds_addr (bytes)
  g0[2] = (unsigned)(ga & 0xFFFFFFFFu);                    // global_addr[31:0]
  g0[3] = (unsigned)((ga >> 32) & 0x01FFFFFFu) | (2u << 30); // ga[56:32] | type=2
  i32x8 g1;
  g1[0] = (int)(3u << 16);                    // workgroup_mask=0, data_size=3 (8B)
  g1[1] = (int)((nelem & 0xFFFFu) << 16);     // tensor_dim0[15:0] @ bits 63:48
  g1[2] = (int)(((nelem >> 16) & 0xFFFFu) | (1u << 16)); // t_dim0[31:16], t_dim1=1
  g1[3] = (int)(nelem << 16);                 // tile_dim0 @ bits 127:112
  g1[4] = 0;                                  // tile_dim1=0, tile_dim2=0 (unused)
  g1[5] = (int)nelem;                         // tensor_dim0_stride[31:0]
  g1[6] = 0;
  g1[7] = 0;
  i32x4 g2 = {0, 0, 0, 0};
  i32x4 g3 = {0, 0, 0, 0};
  i32x8 g4 = {0, 0, 0, 0, 0, 0, 0, 0};
  __builtin_amdgcn_tensor_load_to_lds(g0, g1, g2, g3, g4, 0);
}

// Load A fragment (16x32 f16, ISA layout) for k-tile `kt` from row-major LDS
// tile with row stride `kstride` halves.
__device__ __forceinline__ v16h load_afrag(const _Float16* tile, int kstride,
                                           int kt, int lane) {
  int m  = lane & 15;
  int k0 = (kt << 5) + ((lane >> 4) << 3);
  const _Float16* p = tile + m * kstride + k0;
  v8h lo = *(const v8h*)(p);
  v8h hi = *(const v8h*)(p + 16);
  v16h a;
#pragma unroll
  for (int i = 0; i < 8; ++i) { a[i] = lo[i]; a[i + 8] = hi[i]; }
  return a;
}

// ---------------------------------------------------------------------------
// Block-cooperative GEMM layer: per-wave 16-row A tiles; B panels DMA'd once
// per block into double-buffered LDS by the TDM (wave 0 issues), consumed by
// all waves as ds_load_b128. Chunk = up to 4 k-tiles (4 KB) of one panel.
// `panelBase` points at 2 contiguous panel buffers (offset arithmetic keeps
// the access chain rooted in one __shared__ object => DS, not FLAT).
// ---------------------------------------------------------------------------
template <bool F32OUT>
__device__ __forceinline__ void gemm_staged(
    const _Float16* inTile, int K,
    const _Float16* __restrict__ Wp, const float* __restrict__ bias,
    void* outPtr, int N, bool relu, int lane, int wave,
    _Float16* panelBase, unsigned panelLdsOff) {
  const int Kt = K >> 5, Nt = N >> 4;
  const int chunkK  = (Kt < 4) ? Kt : 4;
  const int nchunks = Kt / chunkK;
  const int total   = Nt * nchunks;
  const unsigned chunkBytes = (unsigned)chunkK << 10;  // chunkK * 32 * 16 * 2

  if (wave == 0)  // prologue: DMA chunk 0 into panel 0
    tdm_copy_to_lds(Wp, panelLdsOff, chunkBytes);

  v8f c;
  for (int ci = 0; ci < total; ++ci) {
    const int nt = ci / nchunks;
    const int kc = ci % nchunks;
    if (wave == 0) {
      if (ci + 1 < total) {
        const int nt2 = (ci + 1) / nchunks, kc2 = (ci + 1) % nchunks;
        const _Float16* src = Wp + ((size_t)(nt2 * Kt + kc2 * chunkK) << 9);
        tdm_copy_to_lds(src, panelLdsOff + (unsigned)(((ci + 1) & 1) * PANEL_BYTES),
                        chunkBytes);
        __builtin_amdgcn_s_wait_tensorcnt(1);   // chunk ci landed
      } else {
        __builtin_amdgcn_s_wait_tensorcnt(0);
      }
    }
    __syncthreads();  // panel ci visible to all waves

    if (kc == 0) {
      float bv = bias[(nt << 4) + (lane & 15)];
#pragma unroll
      for (int i = 0; i < 8; ++i) c[i] = bv;
    }
    const _Float16* pb = panelBase + ((ci & 1) << 11) + (lane << 4);
    for (int k = 0; k < chunkK; ++k) {
      v16h a = load_afrag(inTile, K, kc * chunkK + k, lane);
      v16h b = *(const v16h*)(pb + (k << 9));
      c = __builtin_amdgcn_wmma_f32_16x16x32_f16(false, a, false, b,
                                                 (short)0, c, false, false);
    }
    if (kc == nchunks - 1) {
      int m0 = (lane >> 4) << 3;
      int n  = (nt << 4) + (lane & 15);
#pragma unroll
      for (int i = 0; i < 8; ++i) {
        float v = c[i];
        if (relu) v = v > 0.0f ? v : 0.0f;
        if (F32OUT) ((float*)outPtr)[(m0 + i) * N + n] = v;
        else        ((_Float16*)outPtr)[(m0 + i) * N + n] = (_Float16)v;
      }
    }
    __syncthreads();  // panel ci consumed; safe to overwrite next round
  }
}

// ---------------------------------------------------------------------------
// Fused pipeline: wave = 16-row tile; block = 32 rows sharing TDM-staged B.
// ---------------------------------------------------------------------------
__global__ void __launch_bounds__(WAVES_PER_BLOCK * 32)
surface_vae_kernel(const float* __restrict__ params, const int* __restrict__ stype,
                   const float* __restrict__ type_emb,
                   const float* __restrict__ W_pe, const float* __restrict__ b_pe,
                   const _Float16* __restrict__ We1p, const float* __restrict__ be1,
                   const _Float16* __restrict__ We2p, const float* __restrict__ be2,
                   const _Float16* __restrict__ We3p, const float* __restrict__ be3,
                   const _Float16* __restrict__ We4p, const float* __restrict__ be4,
                   const float* __restrict__ Wpin,  const float* __restrict__ bpin,
                   const float* __restrict__ Wpout, const float* __restrict__ bpout,
                   const _Float16* __restrict__ Wd1p, const float* __restrict__ bd1,
                   const _Float16* __restrict__ Wd2p, const float* __restrict__ bd2,
                   const _Float16* __restrict__ Wd3p, const float* __restrict__ bd3,
                   const float* __restrict__ Wcls, const float* __restrict__ bcls,
                   const float* __restrict__ Wclo, const float* __restrict__ bclo,
                   const float* __restrict__ Wdr,  const float* __restrict__ bdr,
                   float* __restrict__ out, int B) {
  __shared__ __align__(16)  _Float16 sR0[WAVES_PER_BLOCK][16 * 512]; // h1->h3->hd2
  __shared__ __align__(16)  _Float16 sR1[WAVES_PER_BLOCK][16 * 256]; // x->h2->hd1
  __shared__ __align__(16)  float    sAux[WAVES_PER_BLOCK][16 * 32]; // z+xd -> pd
  __shared__ __align__(128) _Float16 sPanel[2 * PANEL_HALVES];       // 2 x 4KB panels

  const int lane = threadIdx.x & 31;
  const int wave = threadIdx.x >> 5;
  const size_t rowBase =
      ((size_t)blockIdx.x * WAVES_PER_BLOCK + wave) * ROWS_PER_WAVE;
  const size_t lastRow = (size_t)B - 1;

  _Float16* r0  = sR0[wave];
  _Float16* r1  = sR1[wave];
  _Float16* xT  = sR1[wave];                       // x: 16x64, front of R1
  float*    zT  = sAux[wave];                      // z: 16x16 f32 (1 KB)
  _Float16* xdT = (_Float16*)(sAux[wave] + 256);   // xd: 16x32 f16 (1 KB)
  float*    pdT = sAux[wave];                      // pd: 16x32 f32 (2 KB)
  _Float16* panelBase = sPanel;
  const unsigned panelLdsOff = (unsigned)(uintptr_t)(void*)sPanel;

  float* reconp = out;
  float* maskp  = out + (size_t)B * 11;
  float* clsp   = out + (size_t)B * 22;
  float* clop   = out + (size_t)B * 27;
  float* zqp    = out + (size_t)B * 29;
  float* idxp   = out + (size_t)B * 45;

  const int raw_dims[5] = {7, 8, 9, 10, 11};

  // ---- Stage 0: per-type input projection + concat(emb) -> x (16x64 f16) ----
  for (int m = 0; m < ROWS_PER_WAVE; ++m) {
    size_t row = rowBase + m; if (row > lastRow) row = lastRow;
    int t = stype[row];
    const float* wrow = W_pe + ((size_t)t * 32 + lane) * 11;
    const float* prow = params + row * 11;
    float acc = b_pe[t * 32 + lane];
#pragma unroll
    for (int r = 0; r < 11; ++r) acc += prow[r] * wrow[r];
    xT[m * 64 + lane] = (_Float16)acc;
    if (lane < 16) xT[m * 64 + 32 + lane] = (_Float16)type_emb[t * 16 + lane];
    else           xT[m * 64 + 32 + lane] = (_Float16)0.0f;
  }
  lds_fence();

  // ---- Encoder ----
  gemm_staged<false>(xT, 64,  We1p, be1, r0, 512, true,  lane, wave, panelBase, panelLdsOff);
  gemm_staged<false>(r0, 512, We2p, be2, r1, 256, true,  lane, wave, panelBase, panelLdsOff);
  gemm_staged<false>(r1, 256, We3p, be3, r0, 128, true,  lane, wave, panelBase, panelLdsOff);
  gemm_staged<true >(r0, 128, We4p, be4, zT, 16,  false, lane, wave, panelBase, panelLdsOff);

  // ---- FSQ + small heads (lane = row, lanes 0..15) ----
  if (lane < 16) {
    const float half_l[4] = {3.5035f, 2.002f, 2.002f, 2.002f};
    const float offs[4]   = {0.5f, 0.0f, 0.0f, 0.0f};
    const float shiftc[4] = {0.14372693f, 0.0f, 0.0f, 0.0f};
    const float half_w[4] = {4.0f, 2.0f, 2.0f, 2.0f};
    const float basis[4]  = {1.0f, 8.0f, 40.0f, 200.0f};

    int m = lane;
    size_t row = rowBase + m; if (row > lastRow) row = lastRow;
    const float* zr = zT + m * 16;
    float codes[4];
    float idxv = 0.0f;
#pragma unroll
    for (int j = 0; j < 4; ++j) {
      float zp = bpin[j];
#pragma unroll
      for (int i = 0; i < 16; ++i) zp += zr[i] * Wpin[i * 4 + j];
      float bounded = tanhf(zp + shiftc[j]) * half_l[j] - offs[j];
      float q = rintf(bounded);            // round-half-even == jnp.round
      codes[j] = q / half_w[j];
      idxv += (q + half_w[j]) * basis[j];
    }
    idxp[row] = idxv;

    float zqrow[16];
#pragma unroll
    for (int i = 0; i < 16; ++i) {
      float v = bpout[i];
#pragma unroll
      for (int j = 0; j < 4; ++j) v += codes[j] * Wpout[j * 16 + i];
      zqrow[i] = v;
      zqp[row * 16 + i] = v;
      xdT[m * 32 + i] = (_Float16)v;       // xd[:, 0:16] = z_q
    }
    int t = stype[row];
#pragma unroll
    for (int i = 0; i < 16; ++i)
      xdT[m * 32 + 16 + i] = (_Float16)type_emb[t * 16 + i];  // xd[:,16:32]=emb
#pragma unroll
    for (int o = 0; o < 5; ++o) {
      float v = bcls[o];
#pragma unroll
      for (int i = 0; i < 16; ++i) v += zqrow[i] * Wcls[i * 5 + o];
      clsp[row * 5 + o] = v;
    }
#pragma unroll
    for (int o = 0; o < 2; ++o) {
      float v = bclo[o];
#pragma unroll
      for (int i = 0; i < 16; ++i) v += zqrow[i] * Wclo[i * 2 + o];
      clop[row * 2 + o] = v;
    }
  }
  lds_fence();

  // ---- Decoder ----
  gemm_staged<false>(xdT, 32,  Wd1p, bd1, r1, 256, true,  lane, wave, panelBase, panelLdsOff);
  gemm_staged<false>(r1,  256, Wd2p, bd2, r0, 512, true,  lane, wave, panelBase, panelLdsOff);
  gemm_staged<true >(r0,  512, Wd3p, bd3, pdT, 32, false, lane, wave, panelBase, panelLdsOff);

  // ---- Per-type output head + mask ----
  for (int m = 0; m < ROWS_PER_WAVE; ++m) {
    size_t row = rowBase + m;
    bool in_range = row < (size_t)B; if (row > lastRow) row = lastRow;
    int t  = stype[row];
    int rd = raw_dims[t];
    if (lane < 11 && in_range) {
      const float* wr = Wdr + ((size_t)t * 11 + lane) * 32;
      float acc = bdr[t * 11 + lane];
#pragma unroll
      for (int p = 0; p < 32; ++p) acc += pdT[m * 32 + p] * wr[p];
      reconp[row * 11 + lane] = acc;
      maskp[row * 11 + lane]  = (lane < rd) ? 1.0f : 0.0f;
    }
  }
}

// ---------------------------------------------------------------------------
// Launch
// ---------------------------------------------------------------------------
extern "C" void kernel_launch(void* const* d_in, const int* in_sizes, int n_in,
                              void* d_out, int out_size, void* d_ws, size_t ws_size,
                              hipStream_t stream) {
  const int B = in_sizes[0] / 11;

  const float* params   = (const float*)d_in[0];
  const int*   stype    = (const int*)d_in[1];
  const float* type_emb = (const float*)d_in[2];
  const float* W_pe     = (const float*)d_in[3];
  const float* b_pe     = (const float*)d_in[4];
  const float* We1 = (const float*)d_in[5];  const float* be1 = (const float*)d_in[6];
  const float* We2 = (const float*)d_in[7];  const float* be2 = (const float*)d_in[8];
  const float* We3 = (const float*)d_in[9];  const float* be3 = (const float*)d_in[10];
  const float* We4 = (const float*)d_in[11]; const float* be4 = (const float*)d_in[12];
  const float* Wpin  = (const float*)d_in[13]; const float* bpin  = (const float*)d_in[14];
  const float* Wpout = (const float*)d_in[15]; const float* bpout = (const float*)d_in[16];
  const float* Wd1 = (const float*)d_in[17]; const float* bd1 = (const float*)d_in[18];
  const float* Wd2 = (const float*)d_in[19]; const float* bd2 = (const float*)d_in[20];
  const float* Wd3 = (const float*)d_in[21]; const float* bd3 = (const float*)d_in[22];
  const float* Wcls = (const float*)d_in[23]; const float* bcls = (const float*)d_in[24];
  const float* Wclo = (const float*)d_in[25]; const float* bclo = (const float*)d_in[26];
  const float* Wdr  = (const float*)d_in[27]; const float* bdr  = (const float*)d_in[28];

  // Workspace layout (in halves)
  _Float16* ws   = (_Float16*)d_ws;
  _Float16* We1p = ws;                 // 64*512   = 32768
  _Float16* We2p = We1p + 32768;       // 512*256  = 131072
  _Float16* We3p = We2p + 131072;      // 256*128  = 32768
  _Float16* We4p = We3p + 32768;       // 128*16   = 2048
  _Float16* Wd1p = We4p + 2048;        // 32*256   = 8192
  _Float16* Wd2p = Wd1p + 8192;        // 256*512  = 131072
  _Float16* Wd3p = Wd2p + 131072;      // 512*32   = 16384

  auto pack = [&](const float* W, _Float16* dst, int K, int Kp, int N) {
    int total = Kp * N;
    pack_weights<<<(total + 255) / 256, 256, 0, stream>>>(W, dst, K, Kp, N);
  };
  pack(We1,  We1p,  48,  64, 512);
  pack(We2,  We2p, 512, 512, 256);
  pack(We3,  We3p, 256, 256, 128);
  pack(We4,  We4p, 128, 128, 16);
  pack(Wd1,  Wd1p,  32,  32, 256);
  pack(Wd2,  Wd2p, 256, 256, 512);
  pack(Wd3,  Wd3p, 512, 512, 32);

  int rowsPerBlock = WAVES_PER_BLOCK * ROWS_PER_WAVE;
  int blocks = (B + rowsPerBlock - 1) / rowsPerBlock;
  surface_vae_kernel<<<blocks, WAVES_PER_BLOCK * 32, 0, stream>>>(
      params, stype, type_emb, W_pe, b_pe,
      We1p, be1, We2p, be2, We3p, be3, We4p, be4,
      Wpin, bpin, Wpout, bpout,
      Wd1p, bd1, Wd2p, bd2, Wd3p, bd3,
      Wcls, bcls, Wclo, bclo, Wdr, bdr,
      (float*)d_out, B);
}
